// QKVAttentionLegacy_46093589021361
// MI455X (gfx1250) — matrix operand order repaired and checked by reading, not compiled, not run on hardware
//
#include <hip/hip_runtime.h>
#include <hip/hip_bf16.h>

typedef __attribute__((ext_vector_type(16))) __bf16       v16bf;
typedef __attribute__((ext_vector_type(8)))  float        v8f;
typedef __attribute__((ext_vector_type(4)))  unsigned int v4u;

namespace {
constexpr int H    = 8;
constexpr int CH   = 64;
constexpr int SEQ  = 2048;
constexpr int QT   = 128;            // queries per workgroup
constexpr int SB   = 64;             // keys per block
constexpr int QSTR = 36;             // LDS row strides in dwords (144B: 16B-aligned, bank-friendly)
constexpr int KSTR = 36;
constexpr int VSTR = 36;
constexpr int NBLK = SEQ / SB;

// 0.125 = 1/sqrt(CH) (both softmax scales folded), * log2(e) so softmax runs in base-2
constexpr float QSCALE = 0.125f * 1.4426950408889634f;

__device__ __forceinline__ unsigned f2bf(float f) {
  unsigned u = __builtin_bit_cast(unsigned, f);
  return (u + 0x7FFFu + ((u >> 16) & 1u)) >> 16;   // round-to-nearest-even
}
__device__ __forceinline__ unsigned pk(float lo, float hi) {
  return f2bf(lo) | (f2bf(hi) << 16);
}
// fast pack: take high 16 bits of each float (truncate) with one v_perm_b32
__device__ __forceinline__ unsigned pk_trunc(float lo, float hi) {
  return __builtin_amdgcn_perm(__builtin_bit_cast(unsigned, hi),
                               __builtin_bit_cast(unsigned, lo), 0x07060302u);
}
__device__ __forceinline__ v16bf frag(v4u a, v4u b) {
  union { v4u q[2]; v16bf v; } u;
  u.q[0] = a; u.q[1] = b;
  return u.v;
}
__device__ __forceinline__ float fexp2(float x) {
#if __has_builtin(__builtin_amdgcn_exp2f)
  return __builtin_amdgcn_exp2f(x);
#else
  return exp2f(x);
#endif
}
} // namespace

__global__ void __launch_bounds__(256)
flash_attn_wmma_bf16(const float* __restrict__ qkv, float* __restrict__ out) {
  // packed-bf16 LDS: Qs[t][c/2], Ks[buf][s][c/2], Vs[buf][c][s/2]
  __shared__ __align__(16) unsigned int QsU[QT * QSTR];
  __shared__ __align__(16) unsigned int KsU[2][SB * KSTR];
  __shared__ __align__(16) unsigned int VsU[2][CH * VSTR];

  const int tid  = threadIdx.x;
  const int wave = tid >> 5;
  const int lane = tid & 31;
  const int tl   = lane & 15;   // column within 16-wide tile
  const int hh   = lane >> 4;   // which 16-lane half

  const int bh = blockIdx.x >> 4;   // batch*head (0..31)
  const int qt = blockIdx.x & 15;   // query tile (0..15)
  const int b  = bh >> 3;
  const int h  = bh & 7;

  const float* qp = qkv + ((size_t)b * (3 * H * CH) + (size_t)h * (3 * CH)) * SEQ;
  const float* kp = qp + (size_t)CH * SEQ;
  const float* vp = qp + (size_t)(2 * CH) * SEQ;
  float*       op = out + ((size_t)b * (H * CH) + (size_t)h * CH) * SEQ;
  const int    t0 = qt * QT;

  // ---- stage Q transposed as packed bf16 pairs along c (scaled, log2e folded) ----
  #pragma unroll
  for (int i = 0; i < 16; ++i) {
    int u = tid + i * 256;               // 0..4095 = 128 t x 32 cpairs
    int t = u & 127, cp = u >> 7;
    float a0 = qp[(size_t)(2 * cp) * SEQ + t0 + t] * QSCALE;
    float a1 = qp[(size_t)(2 * cp + 1) * SEQ + t0 + t] * QSCALE;
    QsU[t * QSTR + cp] = pk(a0, a1);
  }

  // ---- K/V 64-key block staging (double-buffered) ----
  auto stage = [&](int sb, int buf) {
    unsigned int* kb = &KsU[buf][0];
    unsigned int* vb = &VsU[buf][0];
    #pragma unroll
    for (int i = 0; i < 8; ++i) {
      int u = tid + i * 256;             // 0..2047 = 32 cpairs x 64 s
      int s = u & 63, cp = u >> 6;
      float a0 = kp[(size_t)(2 * cp) * SEQ + sb + s];
      float a1 = kp[(size_t)(2 * cp + 1) * SEQ + sb + s];
      kb[s * KSTR + cp] = pk(a0, a1);
    }
    #pragma unroll
    for (int i = 0; i < 8; ++i) {
      int u = tid + i * 256;             // 0..2047 = 64 c x 32 spairs
      int sp = u & 31, c = u >> 5;
      float a0 = vp[(size_t)c * SEQ + sb + 2 * sp];
      float a1 = vp[(size_t)c * SEQ + sb + 2 * sp + 1];
      vb[c * VSTR + sp] = pk(a0, a1);
    }
  };

  stage(0, 0);
  __syncthreads();

  // ---- loop-invariant Q B-fragments: two 8-bf16 runs along c per half (k = 8h+s, 16+8h+s-8) ----
  const v4u* qrow = (const v4u*)(QsU + (wave * 16 + tl) * QSTR);
  v16bf bq[2];
  #pragma unroll
  for (int kc = 0; kc < 2; ++kc)
    bq[kc] = frag(qrow[4 * kc + hh], qrow[4 * kc + hh + 2]);

  // all-ones A fragment: row-sums of P via the matrix pipe (1^T x P^T)
  const v4u onesv = {0x3F803F80u, 0x3F803F80u, 0x3F803F80u, 0x3F803F80u};
  const v16bf ones = frag(onesv, onesv);

  const v8f z8 = {0.f, 0.f, 0.f, 0.f, 0.f, 0.f, 0.f, 0.f};
  v8f acc[4] = {z8, z8, z8, z8};   // O^T tiles: M=c (4x16), N=t (16)
  v8f accS   = z8;                 // softmax denominator (rows replicated)
  float mrow = -3.0e38f;

  for (int blk = 0; blk < NBLK; ++blk) {
    const int cur = blk & 1;
    if (blk + 1 < NBLK) stage((blk + 1) * SB, cur ^ 1);   // overlap with compute

    const unsigned int* kb = &KsU[cur][0];
    const unsigned int* vb = &VsU[cur][0];

    // ---- S^T tiles: D[m=s,n=t] = sum_c K[c,s]*Q[c,t]; 4 s-quadrants x 2 k-steps ----
    v8f st[4] = {z8, z8, z8, z8};
    #pragma unroll
    for (int ts = 0; ts < 4; ++ts) {
      const v4u* kr = (const v4u*)(kb + (ts * 16 + tl) * KSTR);
      #pragma unroll
      for (int kc = 0; kc < 2; ++kc) {
        v16bf a = frag(kr[4 * kc + hh], kr[4 * kc + hh + 2]);
        st[ts] = __builtin_amdgcn_wmma_f32_16x16x32_bf16(
            false, a, false, bq[kc], (short)0, st[ts], false, false);
      }
    }

    // ---- online softmax in base 2 over 64 keys (32 in-lane + partner half) ----
    float lm = -3.0e38f;
    #pragma unroll
    for (int ts = 0; ts < 4; ++ts)
      #pragma unroll
      for (int r = 0; r < 8; ++r) lm = fmaxf(lm, st[ts][r]);
    lm = fmaxf(lm, __shfl_xor(lm, 16, 32));
    float mn   = fmaxf(mrow, lm);
    float corr = fexp2(mrow - mn);
    mrow = mn;
    float p[4][8];
    #pragma unroll
    for (int ts = 0; ts < 4; ++ts)
      #pragma unroll
      for (int r = 0; r < 8; ++r) p[ts][r] = fexp2(st[ts][r] - mn);
    #pragma unroll
    for (int i = 0; i < 4; ++i) acc[i] *= corr;
    accS *= corr;

    // ---- P (D layout) -> B fragments of P^T: one half-wave exchange per k-step ----
    v16bf bp[2];
    #pragma unroll
    for (int ks = 0; ks < 2; ++ks) {
      const float* A = p[2 * ks];       // s-quadrant 2*ks   (K=0..15 of this k-step)
      const float* B = p[2 * ks + 1];   // s-quadrant 2*ks+1 (K=16..31)
      float recv[8];
      #pragma unroll
      for (int r = 0; r < 8; ++r)
        recv[r] = __shfl_xor(hh ? A[r] : B[r], 16, 32);
      unsigned tb[8];
      #pragma unroll
      for (int j = 0; j < 8; ++j) {
        float lo, hi;
        if (j < 4) { lo = hh ? recv[2 * j]     : A[2 * j];
                     hi = hh ? recv[2 * j + 1] : A[2 * j + 1]; }
        else       { int jj = 2 * j - 8;
                     lo = hh ? B[jj]     : recv[jj];
                     hi = hh ? B[jj + 1] : recv[jj + 1]; }
        tb[j] = pk_trunc(lo, hi);
      }
      v4u ta = {tb[0], tb[1], tb[2], tb[3]};
      v4u tc = {tb[4], tb[5], tb[6], tb[7]};
      bp[ks] = frag(ta, tc);
    }

    // ---- denominator: accS += 1^T * P^T (sums the SAME bf16 p as the numerator) ----
    #pragma unroll
    for (int ks = 0; ks < 2; ++ks)
      accS = __builtin_amdgcn_wmma_f32_16x16x32_bf16(
          false, ones, false, bp[ks], (short)0, accS, false, false);

    // ---- O^T += V * P^T: A-frag = two b128 LDS loads along s per k-step ----
    #pragma unroll
    for (int ci = 0; ci < 4; ++ci) {
      const v4u* vr = (const v4u*)(vb + (ci * 16 + tl) * VSTR);
      #pragma unroll
      for (int ks = 0; ks < 2; ++ks) {
        v16bf av = frag(vr[4 * ks + 2 * hh], vr[4 * ks + 2 * hh + 1]);
        acc[ci] = __builtin_amdgcn_wmma_f32_16x16x32_bf16(
            false, av, false, bp[ks], (short)0, acc[ci], false, false);
      }
    }

    __syncthreads();   // staged next buffer ready; current buffer free to overwrite
  }

  // ---- normalize + store: lanes 0..15 contiguous in t -> 64B segments ----
  float inv = 1.0f / accS[0];   // all rows of accS hold the same per-t sum
  #pragma unroll
  for (int ci = 0; ci < 4; ++ci) {
    #pragma unroll
    for (int r = 0; r < 8; ++r) {
      int c = ci * 16 + hh * 8 + r;
      op[(size_t)c * SEQ + t0 + wave * 16 + tl] = acc[ci][r] * inv;
    }
  }
}

extern "C" void kernel_launch(void* const* d_in, const int* in_sizes, int n_in,
                              void* d_out, int out_size, void* d_ws, size_t ws_size,
                              hipStream_t stream) {
  (void)in_sizes; (void)n_in; (void)d_ws; (void)ws_size; (void)out_size;
  const float* qkv = (const float*)d_in[0];
  float*       out = (float*)d_out;
  dim3 grid(32 * 16);   // (BS*H) x (SEQ/QT)
  dim3 block(256);      // 8 wave32 waves
  hipLaunchKernelGGL(flash_attn_wmma_bf16, grid, block, 0, stream, qkv, out);
}
